// QConv2d_64948495450124
// MI455X (gfx1250) — matrix operand order repaired and checked by reading, not compiled, not run on hardware
//
#include <hip/hip_runtime.h>

typedef __attribute__((ext_vector_type(16))) _Float16 v16h;
typedef __attribute__((ext_vector_type(8)))  float    v8f;

#define IMG 256
#define HO  127
#define WO  127
#define NK  8

struct cpx { float re, im; };
__device__ inline cpx cmul(cpx a, cpx b) { return {a.re*b.re - a.im*b.im, a.re*b.im + a.im*b.re}; }
__device__ inline cpx cadd(cpx a, cpx b) { return {a.re + b.re, a.im + b.im}; }

// ---------------------------------------------------------------------------
// Phase 1: build the 8 circuit unitaries U_k (16x16 complex) from the Rot
// angles, then pack into WMMA A-operand layout with ROW DUPLICATION so the
// observable epilogue is fully lane-local (no cross-lane shuffles).
//
// 32 chunks of 16x32 f16 (K=16..31 zero padded). Per kernel k:
//   chunk 4k+0: rows = Re(s_0..7)   (rows 8..15 duplicate rows 0..7)
//   chunk 4k+1: rows = Re(s_8..15)  (dup)
//   chunk 4k+2: rows = Im(s_0..7)   (dup)
//   chunk 4k+3: rows = Im(s_8..15)  (dup)
// A lane layout (contiguous k, as B): lane L<16 -> M=L, halves 0..15 = K0..15;
//                               lane L+16 -> M=L, halves 0..15 = K16..31
//                               padding = 0.
// Apack[(chunk*32 + lane)*16 + h], _Float16, 32 KB in d_ws.
// ---------------------------------------------------------------------------
__global__ void qconv_build_A(const float* __restrict__ w, _Float16* __restrict__ Apack) {
  int k = threadIdx.x;
  if (k >= NK) return;
  float Ur[16][16], Ui[16][16];
  for (int r = 0; r < 16; ++r)
    for (int c = 0; c < 16; ++c) { Ur[r][c] = (r == c) ? 1.f : 0.f; Ui[r][c] = 0.f; }

  // Rot(phi,theta,omega) on qubit q (qubit q <-> index bit 3-q, MSB-first)
  for (int q = 0; q < 4; ++q) {
    float phi = w[(k*4 + q)*3 + 0];
    float th  = w[(k*4 + q)*3 + 1];
    float om  = w[(k*4 + q)*3 + 2];
    float ch = cosf(0.5f*th), sh = sinf(0.5f*th);
    float ap = -0.5f*(phi + om), am = -0.5f*(phi - om);
    cpx ep = {cosf(ap), sinf(ap)};
    cpx em = {cosf(am), sinf(am)};
    cpx G00 = { ep.re*ch,  ep.im*ch};
    cpx G01 = {-em.re*sh,  em.im*sh};   // -conj(em)*s
    cpx G10 = { em.re*sh,  em.im*sh};
    cpx G11 = { ep.re*ch, -ep.im*ch};   //  conj(ep)*c
    int mask = 1 << (3 - q);
    for (int col = 0; col < 16; ++col)
      for (int r0 = 0; r0 < 16; ++r0) {
        if (r0 & mask) continue;
        int r1 = r0 | mask;
        cpx a = {Ur[r0][col], Ui[r0][col]};
        cpx b = {Ur[r1][col], Ui[r1][col]};
        cpx n0 = cadd(cmul(G00, a), cmul(G01, b));
        cpx n1 = cadd(cmul(G10, a), cmul(G11, b));
        Ur[r0][col] = n0.re; Ui[r0][col] = n0.im;
        Ur[r1][col] = n1.re; Ui[r1][col] = n1.im;
      }
  }
  // CNOT chain (0,1)(1,2)(2,3)(3,0): where control bit set, flip target bit.
  const int cm[4] = {8, 4, 2, 1};
  const int tm[4] = {4, 2, 1, 8};
  for (int g = 0; g < 4; ++g)
    for (int r = 0; r < 16; ++r)
      if ((r & cm[g]) && !(r & tm[g])) {
        int r2 = r | tm[g];
        for (int c = 0; c < 16; ++c) {
          float t;
          t = Ur[r][c]; Ur[r][c] = Ur[r2][c]; Ur[r2][c] = t;
          t = Ui[r][c]; Ui[r][c] = Ui[r2][c]; Ui[r2][c] = t;
        }
      }
  // Pack with duplicated rows.
  for (int g = 0; g < 4; ++g) {
    int chunk = 4*k + g;
    int rbase = (g & 1) * 8;                 // s_0..7 vs s_8..15
    for (int L = 0; L < 32; ++L) {
      int m    = (L < 16) ? L : (L - 16);    // M index within chunk
      int srow = rbase + (m & 7);            // rows 8..15 duplicate 0..7
      int kcnt = (L < 16) ? 16 : 0;          // K0..15 in low lanes, pad in high
      int base = (chunk*32 + L) * 16;
      for (int h = 0; h < kcnt; ++h) {
        float v = (g >= 2) ? Ui[srow][h] : Ur[srow][h];
        Apack[base + h] = (_Float16)v;
      }
      for (int h = kcnt; h < 16; ++h) Apack[base + h] = (_Float16)0.f;
    }
  }
}

// ---------------------------------------------------------------------------
// Phase 2: one wave per (b,i) output row; 8 tiles of 16 patches along j.
// Per tile: load+normalize patches into the WMMA B operand, then per kernel
// 4x v_wmma_f32_16x16x32_f16 and a fully lane-local observable epilogue.
// ---------------------------------------------------------------------------
__global__ __launch_bounds__(256) void qconv_main(const float* __restrict__ x,
                                                  const _Float16* __restrict__ Apack,
                                                  float* __restrict__ out) {
  __shared__ __align__(32) _Float16 ldsA[32 * 32 * 16];  // 32 KB packed A
  {
    const float4* src = (const float4*)Apack;
    float4* dst = (float4*)ldsA;
    for (int t = threadIdx.x; t < (32 * 32 * 16) / 8; t += 256) dst[t] = src[t];
  }
  __syncthreads();

  int wave = threadIdx.x >> 5;
  int lane = threadIdx.x & 31;
  int row  = blockIdx.x * 8 + wave;          // 0 .. 64*127-1
  int b    = row / HO;
  int i    = row - b * HO;
  const float* xb = x + (size_t)b * IMG * IMG + (size_t)(i * 2) * IMG;

  for (int jt = 0; jt < 8; ++jt) {
    int j = jt * 16 + (lane & 15);
    bool active = (lane < 16) && (j < WO);

    float v[16];
#pragma unroll
    for (int t = 0; t < 16; ++t) v[t] = 0.f;
    if (active) {
#pragma unroll
      for (int kh = 0; kh < 4; ++kh) {       // each patch row: two b64 loads
        const float2* p = (const float2*)(xb + kh * IMG + j * 2);
        float2 a0 = p[0], a1 = p[1];
        v[kh*4 + 0] = a0.x; v[kh*4 + 1] = a0.y;
        v[kh*4 + 2] = a1.x; v[kh*4 + 3] = a1.y;
      }
    }
    float n2 = 0.f;
#pragma unroll
    for (int t = 0; t < 16; ++t) n2 += v[t] * v[t];
    float inv = 1.0f / (sqrtf(n2) + 1e-12f);

    // B operand: lane L<16 = column N=L, K=0..15; lanes 16..31 = K padding (0)
    v16h bm;
#pragma unroll
    for (int t = 0; t < 16; ++t) bm[t] = (_Float16)(v[t] * inv);

#pragma unroll
    for (int k = 0; k < NK; ++k) {
      v16h a0 = *(const v16h*)&ldsA[((4*k + 0) * 32 + lane) * 16];
      v16h a1 = *(const v16h*)&ldsA[((4*k + 1) * 32 + lane) * 16];
      v16h a2 = *(const v16h*)&ldsA[((4*k + 2) * 32 + lane) * 16];
      v16h a3 = *(const v16h*)&ldsA[((4*k + 3) * 32 + lane) * 16];
      v8f cz = {};
      // Every lane gets, for its patch: d0=Re(s_t), d1=Re(s_{t+8}),
      // d2=Im(s_t), d3=Im(s_{t+8}), t=0..7 -> epilogue needs no shuffles.
      v8f d0 = __builtin_amdgcn_wmma_f32_16x16x32_f16(false, a0, false, bm,
                                                      (short)0, cz, false, false);
      v8f d1 = __builtin_amdgcn_wmma_f32_16x16x32_f16(false, a1, false, bm,
                                                      (short)0, cz, false, false);
      v8f d2 = __builtin_amdgcn_wmma_f32_16x16x32_f16(false, a2, false, bm,
                                                      (short)0, cz, false, false);
      v8f d3 = __builtin_amdgcn_wmma_f32_16x16x32_f16(false, a3, false, bm,
                                                      (short)0, cz, false, false);
      float p0 = 0.f, p1 = 0.f, cR = 0.f, cI = 0.f;
#pragma unroll
      for (int t = 0; t < 8; ++t) {
        float re0 = d0[t], re1 = d1[t], im0 = d2[t], im1 = d3[t];
        p0 += re0 * re0 + im0 * im0;
        p1 += re1 * re1 + im1 * im1;
        cR += re0 * re1 + im0 * im1;   // Re<conj(s0)|s1>
        cI += re0 * im1 - im0 * re1;   // Im<conj(s0)|s1>
      }
      if (active) {
        float* o = out + ((size_t)(b * 24 + 3 * k) * HO + i) * WO + j;
        o[0]                   = 2.f * cR;   // <X0>
        o[(size_t)HO * WO]     = 2.f * cI;   // <Y0>
        o[(size_t)2 * HO * WO] = p0 - p1;    // <Z0>
      }
    }
  }
}

extern "C" void kernel_launch(void* const* d_in, const int* in_sizes, int n_in,
                              void* d_out, int out_size, void* d_ws, size_t ws_size,
                              hipStream_t stream) {
  const float* x = (const float*)d_in[0];          // (64,1,256,256) f32
  const float* w = (const float*)d_in[1];          // (8,4,3) f32
  float* out = (float*)d_out;                      // (64,24,127,127) f32
  _Float16* Apack = (_Float16*)d_ws;               // 32 KB packed A operands

  qconv_build_A<<<1, 8, 0, stream>>>(w, Apack);
  // 64*127 = 8128 output rows, 8 waves/block -> 1016 blocks of 256 threads
  qconv_main<<<1016, 256, 0, stream>>>(x, Apack, out);
}
